// KnowledgeInjectionLayer_48490180772375
// MI455X (gfx1250) — matrix operand order repaired and checked by reading, not compiled, not run on hardware
//
#include <hip/hip_runtime.h>
#include <hip/hip_bf16.h>

// Problem constants (match the reference).
#define OUT_DIM 4096
#define IN_DIM  4096
#define RANK    8
#define BS_DIM  4096          // B * S = 2 * 2048
#define NELEM   (4096u * 4096u)

typedef __attribute__((ext_vector_type(16))) __bf16 v16bf;
typedef __attribute__((ext_vector_type(8)))  __bf16 bf16x8;
typedef __attribute__((ext_vector_type(4)))  __bf16 bf16x4;
typedef __attribute__((ext_vector_type(8)))  float  v8f;

// ---------------------------------------------------------------------------
// Kernel 1: Delta[r][i] = R_new[r][i] - sum_o P[o][r] * W_orig[o][i]
// ---------------------------------------------------------------------------
__global__ void kil_delta_kernel(const float* __restrict__ W,
                                 const float* __restrict__ P,
                                 const float* __restrict__ Rn,
                                 float* __restrict__ Delta) {
    const int i = blockIdx.x * blockDim.x + threadIdx.x;   // 0..IN_DIM-1
    float acc[RANK];
#pragma unroll
    for (int r = 0; r < RANK; ++r) acc[r] = 0.0f;

    for (int o = 0; o < OUT_DIM; ++o) {
        const float w  = W[(size_t)o * IN_DIM + i];
        const float4 p0 = *(const float4*)&P[o * RANK];
        const float4 p1 = *(const float4*)&P[o * RANK + 4];
        acc[0] += p0.x * w;  acc[1] += p0.y * w;
        acc[2] += p0.z * w;  acc[3] += p0.w * w;
        acc[4] += p1.x * w;  acc[5] += p1.y * w;
        acc[6] += p1.z * w;  acc[7] += p1.w * w;
    }
#pragma unroll
    for (int r = 0; r < RANK; ++r)
        Delta[(size_t)r * IN_DIM + i] = Rn[(size_t)r * IN_DIM + i] - acc[r];
}

// ---------------------------------------------------------------------------
// Kernel 2: T[m][r] = sum_k x[m][k] * Delta[r][k]
// ---------------------------------------------------------------------------
__global__ void kil_t_kernel(const float* __restrict__ x,
                             const float* __restrict__ Delta,
                             float* __restrict__ T) {
    const int m   = blockIdx.x;
    const int tid = threadIdx.x;
    __shared__ float red[128][RANK];

    float acc[RANK];
#pragma unroll
    for (int r = 0; r < RANK; ++r) acc[r] = 0.0f;

    for (int k = tid; k < IN_DIM; k += 128) {
        const float xv = x[(size_t)m * IN_DIM + k];
#pragma unroll
        for (int r = 0; r < RANK; ++r)
            acc[r] += xv * Delta[(size_t)r * IN_DIM + k];
    }
#pragma unroll
    for (int r = 0; r < RANK; ++r) red[tid][r] = acc[r];
    __syncthreads();

    if (tid < RANK) {
        float s = 0.0f;
        for (int j = 0; j < 128; ++j) s += red[j][tid];
        T[(size_t)m * RANK + tid] = s;
    }
}

// ---------------------------------------------------------------------------
// Kernel S: split f32 -> (hi, lo) bf16 planes, 4 elements / thread.
// ---------------------------------------------------------------------------
__global__ void kil_split_kernel(const float* __restrict__ src,
                                 __bf16* __restrict__ hi,
                                 __bf16* __restrict__ lo) {
    const size_t i4 = ((size_t)blockIdx.x * blockDim.x + threadIdx.x) * 4;
    const float4 v = *(const float4*)&src[i4];
    const float f[4] = {v.x, v.y, v.z, v.w};
    bf16x4 h, l;
#pragma unroll
    for (int j = 0; j < 4; ++j) {
        const __bf16 hh = (__bf16)f[j];
        h[j] = hh;
        l[j] = (__bf16)(f[j] - (float)hh);
    }
    *(bf16x4*)&hi[i4] = h;
    *(bf16x4*)&lo[i4] = l;
}

// ---------------------------------------------------------------------------
// Fragment loaders from LDS (CDNA5 16-bit A/B VGPR layouts).
// A (16x32, MxK): lane l -> row = l&15; elems 0..7  = K[(l>>4)*8 .. +7]
//                                       elems 8..15 = K[(l>>4)*8+16 .. +7]
// B (32x16, KxN): lane l -> col = l&15; elems 0..7  = K[(l>>4)*8 .. +7], elems 8..15 = K[(l>>4)*8+16 .. +7]
// ---------------------------------------------------------------------------
__device__ __forceinline__ v16bf load_frag_a(const __bf16* rowp, int half) {
    bf16x8 c0 = *(const bf16x8*)(rowp + half * 8);
    bf16x8 c1 = *(const bf16x8*)(rowp + half * 8 + 16);
    v16bf r;
#pragma unroll
    for (int i = 0; i < 8; ++i) { r[i] = c0[i]; r[i + 8] = c1[i]; }
    return r;
}

__device__ __forceinline__ v16bf load_frag_b(const __bf16* rowp, int half) {
    bf16x8 c0 = *(const bf16x8*)(rowp + half * 8);
    bf16x8 c1 = *(const bf16x8*)(rowp + half * 8 + 16);
    v16bf r;
#pragma unroll
    for (int i = 0; i < 8; ++i) { r[i] = c0[i]; r[i + 8] = c1[i]; }
    return r;
}

// ===========================================================================
// FAST PATH: pre-split bf16 planes + async global->LDS double buffering.
// Block tile 128(M) x 256(N), K-slab 32. 8 waves as 2(m) x 4(n), each wave
// 64x64 = 4x4 WMMA fragments -> 48 v_wmma per K-slab vs 32 ds fragment loads.
// ===========================================================================
#define FBLK_M 128
#define FBLK_N 256
#define FBLK_K 32
#define FROW_B 80                 // LDS row stride in bytes (32+8 bf16)
#define FROW_E 40                 // LDS row stride in elements
// plane offsets inside one LDS buffer (elements / bytes)
#define XH_E 0
#define XL_E (128 * FROW_E)
#define WH_E (256 * FROW_E)
#define WL_E (512 * FROW_E)
#define BUF_E (768 * FROW_E)      // 30720 elements
#define BUF_B (BUF_E * 2)         // 61440 bytes
#define LDS_TOTAL_B (2 * BUF_B)   // 122880 bytes

// Issue async copies of one bf16 plane tile (rows x 32 elems) into LDS.
// Each 16B segment is one GLOBAL_LOAD_ASYNC_TO_LDS_B128 (tracked by ASYNCcnt).
__device__ __forceinline__ void async_stage_plane(const __bf16* gplane, int row0,
                                                  int k0, unsigned lds_base_b,
                                                  int tid, int rows) {
    const int passes = rows >> 6;             // 4 segs/row * rows / 256 threads
#pragma unroll
    for (int p = 0; p < 4; ++p) {
        if (p >= passes) break;
        const int idx = p * 256 + tid;
        const int row = idx >> 2;
        const int seg = idx & 3;
        const __bf16* g = gplane + (size_t)(row0 + row) * IN_DIM + k0 + seg * 8;
        const unsigned l = lds_base_b + (unsigned)(row * FROW_B + seg * 16);
        asm volatile("global_load_async_to_lds_b128 %0, %1, off"
                     :: "v"(l), "v"(g) : "memory");
    }
}

__global__ __launch_bounds__(256) void kil_gemm_async_kernel(
    const __bf16* __restrict__ Xhi, const __bf16* __restrict__ Xlo,
    const __bf16* __restrict__ Whi, const __bf16* __restrict__ Wlo,
    const float* __restrict__ bias, const float* __restrict__ P,
    const float* __restrict__ T, float* __restrict__ out) {

    extern __shared__ __bf16 smem[];          // 2 buffers x 4 planes
    const unsigned smem_b = (unsigned)(size_t)&smem[0];   // LDS byte address

    const int tid  = threadIdx.x;
    const int lane = tid & 31;                // wave32
    const int wave = tid >> 5;                // 0..7
    const int wm   = wave >> 2;               // 0..1 -> 64-row slab
    const int wn   = wave & 3;                // 0..3 -> 64-col slab
    const int lr   = lane & 15;
    const int half = lane >> 4;

    const int bm = blockIdx.y * FBLK_M;
    const int bn = blockIdx.x * FBLK_N;

    v8f acc[4][4];
#pragma unroll
    for (int mi = 0; mi < 4; ++mi)
#pragma unroll
        for (int ni = 0; ni < 4; ++ni)
#pragma unroll
            for (int e = 0; e < 8; ++e) acc[mi][ni][e] = 0.0f;

    // Prologue: stage slab 0 into buffer 0 (12 async b128 per thread).
    async_stage_plane(Xhi, bm, 0, smem_b + XH_E * 2, tid, FBLK_M);
    async_stage_plane(Xlo, bm, 0, smem_b + XL_E * 2, tid, FBLK_M);
    async_stage_plane(Whi, bn, 0, smem_b + WH_E * 2, tid, FBLK_N);
    async_stage_plane(Wlo, bn, 0, smem_b + WL_E * 2, tid, FBLK_N);

    const int NK = IN_DIM / FBLK_K;
    for (int kk = 0; kk < NK; ++kk) {
        const int cur = kk & 1;

        // Stage next slab into the other buffer while we compute this one.
        if (kk + 1 < NK) {
            const unsigned nb = smem_b + (unsigned)((1 - cur) * BUF_B);
            const int k1 = (kk + 1) * FBLK_K;
            async_stage_plane(Xhi, bm, k1, nb + XH_E * 2, tid, FBLK_M);
            async_stage_plane(Xlo, bm, k1, nb + XL_E * 2, tid, FBLK_M);
            async_stage_plane(Whi, bn, k1, nb + WH_E * 2, tid, FBLK_N);
            async_stage_plane(Wlo, bn, k1, nb + WL_E * 2, tid, FBLK_N);
            asm volatile("s_wait_asynccnt 0xc" ::: "memory");  // cur slab landed
        } else {
            asm volatile("s_wait_asynccnt 0x0" ::: "memory");
        }
        __syncthreads();   // all waves' cur-slab data visible

        const __bf16* bp = smem + (size_t)cur * BUF_E;
        v16bf Ahi[4], Alo[4], Bhi[4], Blo[4];
#pragma unroll
        for (int mi = 0; mi < 4; ++mi) {
            const int row = wm * 64 + mi * 16 + lr;
            Ahi[mi] = load_frag_a(bp + XH_E + row * FROW_E, half);
            Alo[mi] = load_frag_a(bp + XL_E + row * FROW_E, half);
        }
#pragma unroll
        for (int ni = 0; ni < 4; ++ni) {
            const int row = wn * 64 + ni * 16 + lr;
            Bhi[ni] = load_frag_b(bp + WH_E + row * FROW_E, half);
            Blo[ni] = load_frag_b(bp + WL_E + row * FROW_E, half);
        }

        // bf16x3 compensated product: hi*hi + hi*lo + lo*hi (f32 accumulate).
#pragma unroll
        for (int mi = 0; mi < 4; ++mi)
#pragma unroll
            for (int ni = 0; ni < 4; ++ni) {
                acc[mi][ni] = __builtin_amdgcn_wmma_f32_16x16x32_bf16(
                    false, Ahi[mi], false, Bhi[ni], (short)0, acc[mi][ni], false, false);
                acc[mi][ni] = __builtin_amdgcn_wmma_f32_16x16x32_bf16(
                    false, Ahi[mi], false, Blo[ni], (short)0, acc[mi][ni], false, false);
                acc[mi][ni] = __builtin_amdgcn_wmma_f32_16x16x32_bf16(
                    false, Alo[mi], false, Bhi[ni], (short)0, acc[mi][ni], false, false);
            }
        __syncthreads();   // everyone done reading cur before it is restaged
    }

    // Epilogue: C mapping m = base + v + 8*half, o = base + (lane&15);
    // add bias and the rank-8 correction  sum_r T[m][r] * P[o][r].
    float  bo[4];
    float4 p0[4], p1[4];
    int    ocol[4];
#pragma unroll
    for (int ni = 0; ni < 4; ++ni) {
        ocol[ni] = bn + wn * 64 + ni * 16 + lr;
        bo[ni]   = bias[ocol[ni]];
        p0[ni]   = *(const float4*)&P[ocol[ni] * RANK];
        p1[ni]   = *(const float4*)&P[ocol[ni] * RANK + 4];
    }
#pragma unroll
    for (int mi = 0; mi < 4; ++mi) {
        const int mbase = bm + wm * 64 + mi * 16 + half * 8;
#pragma unroll
        for (int v = 0; v < 8; ++v) {
            const int m = mbase + v;
            const float4 t0 = *(const float4*)&T[(size_t)m * RANK];
            const float4 t1 = *(const float4*)&T[(size_t)m * RANK + 4];
#pragma unroll
            for (int ni = 0; ni < 4; ++ni) {
                const float lrk = t0.x * p0[ni].x + t0.y * p0[ni].y +
                                  t0.z * p0[ni].z + t0.w * p0[ni].w +
                                  t1.x * p1[ni].x + t1.y * p1[ni].y +
                                  t1.z * p1[ni].z + t1.w * p1[ni].w;
                out[(size_t)m * OUT_DIM + ocol[ni]] = acc[mi][ni][v] + bo[ni] + lrk;
            }
        }
    }
}

// ===========================================================================
// FALLBACK: round-1 kernel (in-loop f32->bf16 split), used if ws is small.
// ===========================================================================
#define BLK_M 128
#define BLK_N 128
#define BLK_K 32
#define LDS_STRIDE (BLK_K + 8)

__global__ __launch_bounds__(256) void kil_gemm_kernel(
    const float* __restrict__ x, const float* __restrict__ W,
    const float* __restrict__ bias, const float* __restrict__ P,
    const float* __restrict__ T, float* __restrict__ out) {

    __shared__ __bf16 Xhi[BLK_M][LDS_STRIDE];
    __shared__ __bf16 Xlo[BLK_M][LDS_STRIDE];
    __shared__ __bf16 Whi[BLK_N][LDS_STRIDE];
    __shared__ __bf16 Wlo[BLK_N][LDS_STRIDE];

    const int tid  = threadIdx.x;
    const int lane = tid & 31;
    const int wave = tid >> 5;
    const int wm   = wave >> 1;
    const int wn   = wave & 1;
    const int lr   = lane & 15;
    const int half = lane >> 4;

    const int bm = blockIdx.y * BLK_M;
    const int bn = blockIdx.x * BLK_N;

    v8f acc[2][4];
#pragma unroll
    for (int mi = 0; mi < 2; ++mi)
#pragma unroll
        for (int ni = 0; ni < 4; ++ni)
#pragma unroll
            for (int e = 0; e < 8; ++e) acc[mi][ni][e] = 0.0f;

    for (int kk = 0; kk < IN_DIM / BLK_K; ++kk) {
        const int k0 = kk * BLK_K;
#pragma unroll
        for (int p = 0; p < 4; ++p) {
            const int idx = p * 1024 + tid * 4;
            const int row = idx >> 5;
            const int col = idx & 31;
            const float4 xv = *(const float4*)&x[(size_t)(bm + row) * IN_DIM + k0 + col];
            const float4 wv = *(const float4*)&W[(size_t)(bn + row) * IN_DIM + k0 + col];
            const float xf[4] = {xv.x, xv.y, xv.z, xv.w};
            const float wf[4] = {wv.x, wv.y, wv.z, wv.w};
            bf16x4 xh, xl, wh, wl;
#pragma unroll
            for (int j = 0; j < 4; ++j) {
                const __bf16 h = (__bf16)xf[j];
                xh[j] = h;  xl[j] = (__bf16)(xf[j] - (float)h);
                const __bf16 g = (__bf16)wf[j];
                wh[j] = g;  wl[j] = (__bf16)(wf[j] - (float)g);
            }
            *(bf16x4*)&Xhi[row][col] = xh;
            *(bf16x4*)&Xlo[row][col] = xl;
            *(bf16x4*)&Whi[row][col] = wh;
            *(bf16x4*)&Wlo[row][col] = wl;
        }
        __syncthreads();

        v16bf Ahi[2], Alo[2], Bhi[4], Blo[4];
#pragma unroll
        for (int mi = 0; mi < 2; ++mi) {
            const int row = wm * 32 + mi * 16 + lr;
            Ahi[mi] = load_frag_a(&Xhi[row][0], half);
            Alo[mi] = load_frag_a(&Xlo[row][0], half);
        }
#pragma unroll
        for (int ni = 0; ni < 4; ++ni) {
            const int row = wn * 64 + ni * 16 + lr;
            Bhi[ni] = load_frag_b(&Whi[row][0], half);
            Blo[ni] = load_frag_b(&Wlo[row][0], half);
        }
#pragma unroll
        for (int mi = 0; mi < 2; ++mi)
#pragma unroll
            for (int ni = 0; ni < 4; ++ni) {
                acc[mi][ni] = __builtin_amdgcn_wmma_f32_16x16x32_bf16(
                    false, Ahi[mi], false, Bhi[ni], (short)0, acc[mi][ni], false, false);
                acc[mi][ni] = __builtin_amdgcn_wmma_f32_16x16x32_bf16(
                    false, Ahi[mi], false, Blo[ni], (short)0, acc[mi][ni], false, false);
                acc[mi][ni] = __builtin_amdgcn_wmma_f32_16x16x32_bf16(
                    false, Alo[mi], false, Bhi[ni], (short)0, acc[mi][ni], false, false);
            }
        __syncthreads();
    }

    float  bo[4];
    float4 p0[4], p1[4];
    int    ocol[4];
#pragma unroll
    for (int ni = 0; ni < 4; ++ni) {
        ocol[ni] = bn + wn * 64 + ni * 16 + lr;
        bo[ni]   = bias[ocol[ni]];
        p0[ni]   = *(const float4*)&P[ocol[ni] * RANK];
        p1[ni]   = *(const float4*)&P[ocol[ni] * RANK + 4];
    }
#pragma unroll
    for (int mi = 0; mi < 2; ++mi) {
        const int mbase = bm + wm * 32 + mi * 16 + half * 8;
#pragma unroll
        for (int v = 0; v < 8; ++v) {
            const int m = mbase + v;
            const float4 t0 = *(const float4*)&T[(size_t)m * RANK];
            const float4 t1 = *(const float4*)&T[(size_t)m * RANK + 4];
#pragma unroll
            for (int ni = 0; ni < 4; ++ni) {
                const float lrk = t0.x * p0[ni].x + t0.y * p0[ni].y +
                                  t0.z * p0[ni].z + t0.w * p0[ni].w +
                                  t1.x * p1[ni].x + t1.y * p1[ni].y +
                                  t1.z * p1[ni].z + t1.w * p1[ni].w;
                out[(size_t)m * OUT_DIM + ocol[ni]] = acc[mi][ni][v] + bo[ni] + lrk;
            }
        }
    }
}

// ---------------------------------------------------------------------------
// Launch. ws layout: Delta (128KB) | T (128KB) | Xhi | Xlo | Whi | Wlo (32MB ea)
// ---------------------------------------------------------------------------
extern "C" void kernel_launch(void* const* d_in, const int* in_sizes, int n_in,
                              void* d_out, int out_size, void* d_ws, size_t ws_size,
                              hipStream_t stream) {
    const float* x  = (const float*)d_in[0];   // [B*S, IN]
    const float* W  = (const float*)d_in[1];   // [OUT, IN]
    const float* b  = (const float*)d_in[2];   // [OUT]
    const float* P  = (const float*)d_in[3];   // [OUT, RANK]
    const float* Rn = (const float*)d_in[4];   // [RANK, IN]
    float* out = (float*)d_out;

    char*  wsb   = (char*)d_ws;
    float* Delta = (float*)wsb;
    float* T     = (float*)(wsb + (128 << 10));

    kil_delta_kernel<<<IN_DIM / 256, 256, 0, stream>>>(W, P, Rn, Delta);
    kil_t_kernel<<<BS_DIM, 128, 0, stream>>>(x, Delta, T);

    const size_t need = (size_t)(256 << 10) + 4ull * NELEM * sizeof(__bf16);
    if (ws_size >= need) {
        __bf16* Xhi = (__bf16*)(wsb + (256 << 10));
        __bf16* Xlo = Xhi + (size_t)NELEM;
        __bf16* Whi = Xlo + (size_t)NELEM;
        __bf16* Wlo = Whi + (size_t)NELEM;

        kil_split_kernel<<<NELEM / (256 * 4), 256, 0, stream>>>(x, Xhi, Xlo);
        kil_split_kernel<<<NELEM / (256 * 4), 256, 0, stream>>>(W, Whi, Wlo);

        dim3 grid(OUT_DIM / FBLK_N, BS_DIM / FBLK_M);
        kil_gemm_async_kernel<<<grid, 256, LDS_TOTAL_B, stream>>>(
            Xhi, Xlo, Whi, Wlo, b, P, T, out);
    } else {
        dim3 grid(OUT_DIM / BLK_N, BS_DIM / BLK_M);
        kil_gemm_kernel<<<grid, 256, 0, stream>>>(x, W, b, P, T, out);
    }
}
